// MultiScaleMambaEncoder_45372034514954
// MI455X (gfx1250) — matrix operand order repaired and checked by reading, not compiled, not run on hardware
//
#include <hip/hip_runtime.h>
#include <hip/hip_bf16.h>

typedef __attribute__((ext_vector_type(16))) _Float16 v16h;
typedef __attribute__((ext_vector_type(8)))  _Float16 v8h;
typedef __attribute__((ext_vector_type(8)))  float    v8f;

namespace {
constexpr int kB   = 2;
constexpr int kL   = 1024;
constexpr int kDM  = 768;
constexpr int kNL  = 2;
constexpr int kDI  = 1536;          // EXP * DM
constexpr int kDTR = 48;
constexpr int kDTRp = 64;           // DTR padded to WMMA K granularity
constexpr int kNT  = kB * kL;       // 2048 token rows
constexpr int kSS  = 16, kSL = 32;
constexpr int kKS  = 3,  kKL = 9;
} // namespace

// ---------------------------------------------------------------------------
// fp32 -> f16 weight conversion with K padding and N-row padding (zeros).
// src: (rowsValid x K) row-major fp32 ; dst: (rowsTotal x Kp) row-major f16
// ---------------------------------------------------------------------------
__global__ __launch_bounds__(256) void cvt_f16_pad_kernel(
    const float* __restrict__ src, _Float16* __restrict__ dst,
    int K, int Kp, long rowsValid, long total /* rowsTotal*Kp */)
{
  long idx = (long)blockIdx.x * blockDim.x + threadIdx.x;
  if (idx >= total) return;
  int  c = (int)(idx % Kp);
  long r = idx / Kp;
  dst[idx] = (c < K && r < rowsValid) ? (_Float16)src[r * K + c] : (_Float16)0.f;
}

// ---------------------------------------------------------------------------
// RMSNorm: xn = x * rsqrt(mean(x^2)+1e-5) * norm_w  -> f16 (one block per row)
// ---------------------------------------------------------------------------
__global__ __launch_bounds__(256) void rmsnorm_f16_kernel(
    const float* __restrict__ x, const float* __restrict__ nw,
    _Float16* __restrict__ o, int n)
{
  __shared__ float red[8];
  __shared__ float scale_sh;
  const int row = blockIdx.x;
  const float* xr = x + (size_t)row * n;
  float ss = 0.f;
  for (int i = threadIdx.x; i < n; i += 256) { float v = xr[i]; ss += v * v; }
  for (int off = 16; off > 0; off >>= 1) ss += __shfl_down(ss, off, 32);
  const int wave = threadIdx.x >> 5, lane = threadIdx.x & 31;
  if (lane == 0) red[wave] = ss;
  __syncthreads();
  if (threadIdx.x == 0) {
    float t = 0.f;
    for (int w = 0; w < 8; ++w) t += red[w];
    scale_sh = rsqrtf(t / (float)n + 1e-5f);
  }
  __syncthreads();
  const float sc = scale_sh;
  _Float16* orow = o + (size_t)row * n;
  for (int i = threadIdx.x; i < n; i += 256)
    orow[i] = (_Float16)(xr[i] * sc * nw[i]);
}

// ---------------------------------------------------------------------------
// WMMA f16 GEMM:  C(MxN,f32) = A(MxKp,f16) * W(NpxKp,f16)^T [+bias][+resid]
//
// Wave tile: 16M x 64N = four 16x16 accumulator chains sharing each A
// fragment (4 matrix ops per A load, independent D regs -> no WMMA hazards).
// Block: 8 waves stacked along M -> 128M x 64N block tile; all 8 waves
// stream the same four W fragments (high L0/L2 reuse).
// K loop unrolled x2 with two disjoint fragment register sets (no rotation
// copies); odd-step preload is unconditional since Kp % 64 == 0.
// Requires: M % 128 == 0, Np % 64 == 0 (W zero-padded rows), Kp % 64 == 0,
//           N % 16 == 0 (store guard per 16-col subtile).
// ---------------------------------------------------------------------------
__global__ __launch_bounds__(256) void gemm_f16_wmma_kernel(
    const _Float16* __restrict__ A, const _Float16* __restrict__ W,
    const float* __restrict__ bias, const float* __restrict__ resid,
    float* __restrict__ C, int M, int N, int Np, int Kp)
{
  const int lane = threadIdx.x & 31;
  const int wave = threadIdx.x >> 5;
  const int m0 = blockIdx.y * 128 + wave * 16;
  const int n0 = blockIdx.x * 64;
  const int half = lane >> 4;              // 0/1: which K-half this lane holds
  const int l15  = lane & 15;

  const _Float16* arow = A + (size_t)(m0 + l15) * Kp;
  const _Float16* wr0 = W + (size_t)(n0 +  0 + l15) * Kp;
  const _Float16* wr1 = W + (size_t)(n0 + 16 + l15) * Kp;
  const _Float16* wr2 = W + (size_t)(n0 + 32 + l15) * Kp;
  const _Float16* wr3 = W + (size_t)(n0 + 48 + l15) * Kp;

  // A fragment (16x32): elems 0..15 = K kk+16h..+15 (contiguous, half-split)
  auto load_a = [&](int kk) {
    return *(const v16h*)(arow + kk + half * 16);
  };
  // B fragment (32x16, B = W^T): 16 contiguous K of weight row n, half-split
  auto load_b = [&](const _Float16* wr, int kk) {
    return *(const v16h*)(wr + kk + half * 16);
  };

  v8f acc0 = {}, acc1 = {}, acc2 = {}, acc3 = {};

  // even-set prologue (k = 0)
  v16h aE  = load_a(0);
  v16h bE0 = load_b(wr0, 0), bE1 = load_b(wr1, 0);
  v16h bE2 = load_b(wr2, 0), bE3 = load_b(wr3, 0);

  for (int k = 0; k < Kp; k += 64) {
    // odd-set preload (k+32) -- unconditional, Kp % 64 == 0
    v16h aO  = load_a(k + 32);
    v16h bO0 = load_b(wr0, k + 32), bO1 = load_b(wr1, k + 32);
    v16h bO2 = load_b(wr2, k + 32), bO3 = load_b(wr3, k + 32);

    acc0 = __builtin_amdgcn_wmma_f32_16x16x32_f16(false, aE, false, bE0, (short)0, acc0, false, false);
    acc1 = __builtin_amdgcn_wmma_f32_16x16x32_f16(false, aE, false, bE1, (short)0, acc1, false, false);
    acc2 = __builtin_amdgcn_wmma_f32_16x16x32_f16(false, aE, false, bE2, (short)0, acc2, false, false);
    acc3 = __builtin_amdgcn_wmma_f32_16x16x32_f16(false, aE, false, bE3, (short)0, acc3, false, false);

    if (k + 64 < Kp) {                     // even-set preload (k+64)
      aE  = load_a(k + 64);
      bE0 = load_b(wr0, k + 64); bE1 = load_b(wr1, k + 64);
      bE2 = load_b(wr2, k + 64); bE3 = load_b(wr3, k + 64);
    }

    acc0 = __builtin_amdgcn_wmma_f32_16x16x32_f16(false, aO, false, bO0, (short)0, acc0, false, false);
    acc1 = __builtin_amdgcn_wmma_f32_16x16x32_f16(false, aO, false, bO1, (short)0, acc1, false, false);
    acc2 = __builtin_amdgcn_wmma_f32_16x16x32_f16(false, aO, false, bO2, (short)0, acc2, false, false);
    acc3 = __builtin_amdgcn_wmma_f32_16x16x32_f16(false, aO, false, bO3, (short)0, acc3, false, false);
  }

  // D layout: VGPR r -> row m0 + 8*half + r, col nbase + l15
  const int mr = m0 + half * 8;
  const v8f accs[4] = {acc0, acc1, acc2, acc3};
#pragma unroll
  for (int sub = 0; sub < 4; ++sub) {
    const int nbase = n0 + sub * 16;
    if (nbase >= N) continue;              // wave-uniform (N % 16 == 0)
    const int col = nbase + l15;
    const float bv = bias ? bias[col] : 0.f;
#pragma unroll
    for (int r = 0; r < 8; ++r) {
      size_t idx = (size_t)(mr + r) * N + col;
      float v = accs[sub][r] + bv;
      if (resid) v += resid[idx];
      C[idx] = v;
    }
  }
}

// ---------------------------------------------------------------------------
// Depthwise causal conv1d (+bias) over L followed by SiLU.
// Input xi = xz[:, :DI] (row stride 2*DI). Writes fp32 + f16 copies.
// ---------------------------------------------------------------------------
template <int KC>
__global__ __launch_bounds__(256) void conv_silu_kernel(
    const float* __restrict__ xz, const float* __restrict__ cw,
    const float* __restrict__ cb, float* __restrict__ xi_act,
    _Float16* __restrict__ xi16)
{
  int idx = blockIdx.x * 256 + threadIdx.x;      // over B*L*DI
  if (idx >= kNT * kDI) return;
  const int d = idx % kDI;
  const int t = (idx / kDI) % kL;
  const int b = idx / (kDI * kL);
  float acc = cb[d];
#pragma unroll
  for (int j = 0; j < KC; ++j) {
    int tt = t - (KC - 1) + j;
    if (tt >= 0)
      acc += cw[d * KC + j] * xz[(size_t)(b * kL + tt) * (2 * kDI) + d];
  }
  float s = acc / (1.f + __expf(-acc));          // SiLU
  xi_act[idx] = s;
  xi16[idx]   = (_Float16)s;
}

// ---------------------------------------------------------------------------
// Pack dbc[:, :DTR] into f16 (NT x 64) with zero pad for the dt GEMM.
// ---------------------------------------------------------------------------
__global__ __launch_bounds__(256) void pack_dt_kernel(
    const float* __restrict__ dbc, int nw, _Float16* __restrict__ a16)
{
  int idx = blockIdx.x * 256 + threadIdx.x;      // over NT*64
  if (idx >= kNT * kDTRp) return;
  const int c = idx & (kDTRp - 1);
  const int r = idx >> 6;
  a16[idx] = (c < kDTR) ? (_Float16)dbc[(size_t)r * nw + c] : (_Float16)0.f;
}

// ---------------------------------------------------------------------------
// Selective scan. One thread per (batch,channel). S states in registers.
// Fuses softplus(delta), y = scan + x*D, and the SiLU(z) gate. Emits f16 y.
// grid = (DI/256, B); B_t/C_t staged through LDS once per timestep.
// ---------------------------------------------------------------------------
template <int S>
__global__ __launch_bounds__(256) void scan_kernel(
    const float* __restrict__ deltapre,   // NT x DI
    const float* __restrict__ dbc, int nw,// NT x nw (B at DTR, C at DTR+S)
    const float* __restrict__ xi_act,     // NT x DI
    const float* __restrict__ xz,         // NT x 2*DI (z at col DI+d)
    const float* __restrict__ A_log,      // DI x S
    const float* __restrict__ Dp,         // DI
    _Float16* __restrict__ y16)           // NT x DI
{
  __shared__ float sB[S], sC[S];
  const int d = blockIdx.x * 256 + threadIdx.x;
  const int b = blockIdx.y;
  float A[S], h[S];
#pragma unroll
  for (int s = 0; s < S; ++s) { A[s] = -__expf(A_log[(size_t)d * S + s]); h[s] = 0.f; }
  const float Dd = Dp[d];
  for (int t = 0; t < kL; ++t) {
    const size_t row = (size_t)(b * kL + t);
    if (threadIdx.x < 2 * S) {
      float v = dbc[row * nw + kDTR + threadIdx.x];
      if (threadIdx.x < S) sB[threadIdx.x] = v; else sC[threadIdx.x - S] = v;
    }
    __syncthreads();
    float dp = deltapre[row * kDI + d];
    float dt = (dp > 20.f) ? dp : log1pf(__expf(dp));   // softplus
    float xt = xi_act[row * kDI + d];
    float dx = dt * xt;
    float y = 0.f;
#pragma unroll
    for (int s = 0; s < S; ++s) {
      h[s] = __expf(dt * A[s]) * h[s] + dx * sB[s];
      y += h[s] * sC[s];
    }
    float z = xz[row * (2 * kDI) + kDI + d];
    float g = z / (1.f + __expf(-z));                   // SiLU(z)
    y16[row * kDI + d] = (_Float16)((y + xt * Dd) * g);
    __syncthreads();
  }
}

// ---------------------------------------------------------------------------
// xc16 = concat(xs, xl) as f16 (NT x 2*DM)
// ---------------------------------------------------------------------------
__global__ __launch_bounds__(256) void concat_f16_kernel(
    const float* __restrict__ xs, const float* __restrict__ xl,
    _Float16* __restrict__ xc16)
{
  int idx = blockIdx.x * 256 + threadIdx.x;      // over NT*2*DM
  if (idx >= kNT * 2 * kDM) return;
  const int c = idx % (2 * kDM);
  const int r = idx / (2 * kDM);
  float v = (c < kDM) ? xs[(size_t)r * kDM + c] : xl[(size_t)r * kDM + (c - kDM)];
  xc16[idx] = (_Float16)v;
}

// ---------------------------------------------------------------------------
// out = LN( sigmoid(gatepre)*xs + (1-sigmoid)*xl + projpre ). One block/row.
// ---------------------------------------------------------------------------
__global__ __launch_bounds__(256) void fuse_ln_kernel(
    const float* __restrict__ gatepre, const float* __restrict__ projpre,
    const float* __restrict__ xs, const float* __restrict__ xl,
    const float* __restrict__ lnw, const float* __restrict__ lnb,
    float* __restrict__ out)
{
  __shared__ float sh[kDM];
  __shared__ float red[16];
  __shared__ float stats[2];
  const int row = blockIdx.x;
  const size_t base = (size_t)row * kDM;
  float s1 = 0.f, s2 = 0.f;
  for (int i = threadIdx.x; i < kDM; i += 256) {
    float g = 1.f / (1.f + __expf(-gatepre[base + i]));
    float o = g * xs[base + i] + (1.f - g) * xl[base + i] + projpre[base + i];
    sh[i] = o; s1 += o; s2 += o * o;
  }
  for (int off = 16; off > 0; off >>= 1) {
    s1 += __shfl_down(s1, off, 32);
    s2 += __shfl_down(s2, off, 32);
  }
  const int wave = threadIdx.x >> 5, lane = threadIdx.x & 31;
  if (lane == 0) { red[wave] = s1; red[8 + wave] = s2; }
  __syncthreads();
  if (threadIdx.x == 0) {
    float a = 0.f, b = 0.f;
    for (int w = 0; w < 8; ++w) { a += red[w]; b += red[8 + w]; }
    float mu = a / kDM;
    stats[0] = mu;
    stats[1] = rsqrtf(b / kDM - mu * mu + 1e-5f);
  }
  __syncthreads();
  const float mu = stats[0], inv = stats[1];
  for (int i = threadIdx.x; i < kDM; i += 256)
    out[base + i] = (sh[i] - mu) * inv * lnw[i] + lnb[i];
}

// ---------------------------------------------------------------------------
// Host-side helpers
// ---------------------------------------------------------------------------
namespace {

inline char* bump(char*& p, size_t bytes) {
  char* r = p;
  p += (bytes + 255) & ~(size_t)255;
  return r;
}

inline void launch_gemm(const _Float16* A, const _Float16* W, const float* bias,
                        const float* resid, float* C, int M, int N, int Np,
                        int Kp, hipStream_t st) {
  dim3 grid(Np / 64, M / 128);
  gemm_f16_wmma_kernel<<<grid, 256, 0, st>>>(A, W, bias, resid, C, M, N, Np, Kp);
}

inline void launch_cvt(const float* src, _Float16* dst, int K, int Kp,
                       long rowsValid, long rowsTotal, hipStream_t st) {
  long total = rowsTotal * (long)Kp;
  cvt_f16_pad_kernel<<<(unsigned)((total + 255) / 256), 256, 0, st>>>(
      src, dst, K, Kp, rowsValid, total);
}

struct BranchW {
  const float *conv_w, *conv_b, *dt_b, *A_log, *D, *norm_w;
  const _Float16 *in_w16, *x_w16, *dt_w16, *out_w16;
};

struct Scratch {
  _Float16 *xn16, *xi16, *dtA16, *y16;
  float *xz, *xi_act, *dbc, *deltapre;
};

// Runs NL mamba layers of one branch. cur/nxt are NTxDM fp32 ping-pong
// buffers; returns pointer holding the final activations.
template <int S, int KC>
float* run_branch(const BranchW& w, float* cur, float* nxt, const Scratch& sc,
                  hipStream_t st) {
  const int nw  = kDTR + 2 * S;        // x_w logical rows (80 / 112)
  const int nwP = (nw + 63) & ~63;     // zero-padded rows (128 / 128)
  for (int i = 0; i < kNL; ++i) {
    rmsnorm_f16_kernel<<<kNT, 256, 0, st>>>(cur, w.norm_w + (size_t)i * kDM,
                                            sc.xn16, kDM);
    // xz = xn @ in_w^T  (NT x 2*DI)
    launch_gemm(sc.xn16, w.in_w16 + (size_t)i * 2 * kDI * kDM, nullptr, nullptr,
                sc.xz, kNT, 2 * kDI, 2 * kDI, kDM, st);
    // depthwise conv + SiLU
    conv_silu_kernel<KC><<<(kNT * kDI + 255) / 256, 256, 0, st>>>(
        sc.xz, w.conv_w + (size_t)i * kDI * KC, w.conv_b + (size_t)i * kDI,
        sc.xi_act, sc.xi16);
    // dbc = xi @ x_w^T  (NT x nw, W rows padded to nwP)
    launch_gemm(sc.xi16, w.x_w16 + (size_t)i * nwP * kDI, nullptr, nullptr,
                sc.dbc, kNT, nw, nwP, kDI, st);
    // deltapre = dbc[:, :DTR] @ dt_w^T + dt_b  (NT x DI)
    pack_dt_kernel<<<(kNT * kDTRp + 255) / 256, 256, 0, st>>>(sc.dbc, nw, sc.dtA16);
    launch_gemm(sc.dtA16, w.dt_w16 + (size_t)i * kDI * kDTRp,
                w.dt_b + (size_t)i * kDI, nullptr, sc.deltapre, kNT, kDI, kDI,
                kDTRp, st);
    // selective scan (fuses softplus, +x*D, SiLU(z) gate) -> y16
    scan_kernel<S><<<dim3(kDI / 256, kB), 256, 0, st>>>(
        sc.deltapre, sc.dbc, nw, sc.xi_act, sc.xz,
        w.A_log + (size_t)i * kDI * S, w.D + (size_t)i * kDI, sc.y16);
    // nxt = cur + y @ out_w^T  (residual fused into GEMM store)
    launch_gemm(sc.y16, w.out_w16 + (size_t)i * kDM * kDI, nullptr, cur, nxt,
                kNT, kDM, kDM, kDI, st);
    float* t = cur; cur = nxt; nxt = t;
  }
  return cur;
}

} // namespace

// ---------------------------------------------------------------------------
// Entry point
// ---------------------------------------------------------------------------
extern "C" void kernel_launch(void* const* d_in, const int* in_sizes, int n_in,
                              void* d_out, int out_size, void* d_ws, size_t ws_size,
                              hipStream_t stream) {
  (void)in_sizes; (void)n_in; (void)out_size; (void)ws_size;
  const float* x = (const float*)d_in[0];
  const float* s_in_w   = (const float*)d_in[1];
  const float* s_conv_w = (const float*)d_in[2];
  const float* s_conv_b = (const float*)d_in[3];
  const float* s_x_w    = (const float*)d_in[4];
  const float* s_dt_w   = (const float*)d_in[5];
  const float* s_dt_b   = (const float*)d_in[6];
  const float* s_A_log  = (const float*)d_in[7];
  const float* s_D      = (const float*)d_in[8];
  const float* s_out_w  = (const float*)d_in[9];
  const float* s_norm_w = (const float*)d_in[10];
  const float* l_in_w   = (const float*)d_in[11];
  const float* l_conv_w = (const float*)d_in[12];
  const float* l_conv_b = (const float*)d_in[13];
  const float* l_x_w    = (const float*)d_in[14];
  const float* l_dt_w   = (const float*)d_in[15];
  const float* l_dt_b   = (const float*)d_in[16];
  const float* l_A_log  = (const float*)d_in[17];
  const float* l_D      = (const float*)d_in[18];
  const float* l_out_w  = (const float*)d_in[19];
  const float* l_norm_w = (const float*)d_in[20];
  const float* gate_w   = (const float*)d_in[21];
  const float* gate_b   = (const float*)d_in[22];
  const float* proj_w   = (const float*)d_in[23];
  const float* proj_b   = (const float*)d_in[24];
  const float* ln_w     = (const float*)d_in[25];
  const float* ln_b     = (const float*)d_in[26];
  float* out = (float*)d_out;

  const int nwS = kDTR + 2 * kSS, nwSP = 128;  // 80  -> 128 (64-mult)
  const int nwL = kDTR + 2 * kSL, nwLP = 128;  // 112 -> 128 (64-mult)

  // ---- carve scratch -------------------------------------------------------
  char* p = (char*)d_ws;
  _Float16* s_in_w16  = (_Float16*)bump(p, (size_t)kNL * 2 * kDI * kDM * 2);
  _Float16* l_in_w16  = (_Float16*)bump(p, (size_t)kNL * 2 * kDI * kDM * 2);
  _Float16* s_x_w16   = (_Float16*)bump(p, (size_t)kNL * nwSP * kDI * 2);
  _Float16* l_x_w16   = (_Float16*)bump(p, (size_t)kNL * nwLP * kDI * 2);
  _Float16* s_dt_w16  = (_Float16*)bump(p, (size_t)kNL * kDI * kDTRp * 2);
  _Float16* l_dt_w16  = (_Float16*)bump(p, (size_t)kNL * kDI * kDTRp * 2);
  _Float16* s_out_w16 = (_Float16*)bump(p, (size_t)kNL * kDM * kDI * 2);
  _Float16* l_out_w16 = (_Float16*)bump(p, (size_t)kNL * kDM * kDI * 2);
  _Float16* gate_w16  = (_Float16*)bump(p, (size_t)kDM * 2 * kDM * 2);
  _Float16* proj_w16  = (_Float16*)bump(p, (size_t)kDM * 2 * kDM * 2);
  float* xs   = (float*)bump(p, (size_t)kNT * kDM * 4);
  float* xl   = (float*)bump(p, (size_t)kNT * kDM * 4);
  float* tmpS = (float*)bump(p, (size_t)kNT * kDM * 4);
  float* tmpL = (float*)bump(p, (size_t)kNT * kDM * 4);
  Scratch sc;
  sc.xn16     = (_Float16*)bump(p, (size_t)kNT * kDM * 2);
  sc.xi16     = (_Float16*)bump(p, (size_t)kNT * kDI * 2);
  sc.dtA16    = (_Float16*)bump(p, (size_t)kNT * kDTRp * 2);
  sc.y16      = (_Float16*)bump(p, (size_t)kNT * kDI * 2);
  sc.xz       = (float*)bump(p, (size_t)kNT * 2 * kDI * 4);
  sc.xi_act   = (float*)bump(p, (size_t)kNT * kDI * 4);
  sc.dbc      = (float*)bump(p, (size_t)kNT * nwL * 4);
  sc.deltapre = (float*)bump(p, (size_t)kNT * kDI * 4);
  _Float16* xc16   = (_Float16*)bump(p, (size_t)kNT * 2 * kDM * 2);
  float* gatepre   = (float*)bump(p, (size_t)kNT * kDM * 4);
  float* projpre   = (float*)bump(p, (size_t)kNT * kDM * 4);

  // ---- convert weights to f16 (K-padded / N-row-padded where needed) -------
  launch_cvt(s_in_w,  s_in_w16,  kDM, kDM, (long)kNL * 2 * kDI, (long)kNL * 2 * kDI, stream);
  launch_cvt(l_in_w,  l_in_w16,  kDM, kDM, (long)kNL * 2 * kDI, (long)kNL * 2 * kDI, stream);
  for (int i = 0; i < kNL; ++i) {   // x_w: per-layer so row padding stays per-layer
    launch_cvt(s_x_w + (size_t)i * nwS * kDI, s_x_w16 + (size_t)i * nwSP * kDI,
               kDI, kDI, nwS, nwSP, stream);
    launch_cvt(l_x_w + (size_t)i * nwL * kDI, l_x_w16 + (size_t)i * nwLP * kDI,
               kDI, kDI, nwL, nwLP, stream);
  }
  launch_cvt(s_dt_w,  s_dt_w16,  kDTR, kDTRp, (long)kNL * kDI, (long)kNL * kDI, stream);
  launch_cvt(l_dt_w,  l_dt_w16,  kDTR, kDTRp, (long)kNL * kDI, (long)kNL * kDI, stream);
  launch_cvt(s_out_w, s_out_w16, kDI, kDI, (long)kNL * kDM, (long)kNL * kDM, stream);
  launch_cvt(l_out_w, l_out_w16, kDI, kDI, (long)kNL * kDM, (long)kNL * kDM, stream);
  launch_cvt(gate_w,  gate_w16,  2 * kDM, 2 * kDM, (long)kDM, (long)kDM, stream);
  launch_cvt(proj_w,  proj_w16,  2 * kDM, 2 * kDM, (long)kDM, (long)kDM, stream);

  // ---- initialize both residual streams from x -----------------------------
  hipMemcpyAsync(xs, x, (size_t)kNT * kDM * 4, hipMemcpyDeviceToDevice, stream);
  hipMemcpyAsync(xl, x, (size_t)kNT * kDM * 4, hipMemcpyDeviceToDevice, stream);

  // ---- run both branches ---------------------------------------------------
  BranchW wS{s_conv_w, s_conv_b, s_dt_b, s_A_log, s_D, s_norm_w,
             s_in_w16, s_x_w16, s_dt_w16, s_out_w16};
  BranchW wL{l_conv_w, l_conv_b, l_dt_b, l_A_log, l_D, l_norm_w,
             l_in_w16, l_x_w16, l_dt_w16, l_out_w16};
  float* xsf = run_branch<kSS, kKS>(wS, xs, tmpS, sc, stream);
  float* xlf = run_branch<kSL, kKL>(wL, xl, tmpL, sc, stream);

  // ---- fusion head: gate / proj / LayerNorm --------------------------------
  concat_f16_kernel<<<(kNT * 2 * kDM + 255) / 256, 256, 0, stream>>>(xsf, xlf, xc16);
  launch_gemm(xc16, gate_w16, gate_b, nullptr, gatepre, kNT, kDM, kDM, 2 * kDM, stream);
  launch_gemm(xc16, proj_w16, proj_b, nullptr, projpre, kNT, kDM, kDM, 2 * kDM, stream);
  fuse_ln_kernel<<<kNT, 256, 0, stream>>>(gatepre, projpre, xsf, xlf, ln_w, ln_b, out);
}
